// attention_action_decoder_55851754717642
// MI455X (gfx1250) — matrix operand order repaired and checked by reading, not compiled, not run on hardware
//
#include <hip/hip_runtime.h>
#include <hip/hip_bf16.h>
#include <cstdint>

// ---------------------------------------------------------------------------
// Types for CDNA5 WMMA (wave32): 16x16x32 bf16 -> f32
// ---------------------------------------------------------------------------
typedef __attribute__((ext_vector_type(16))) __bf16 v16bf;
typedef __attribute__((ext_vector_type(8)))  float  v8f;
typedef __attribute__((ext_vector_type(4)))  unsigned int v4u_;
typedef __attribute__((ext_vector_type(8)))  int v8i_;
typedef __attribute__((ext_vector_type(4)))  int v4i_;

union FragAB { v16bf v; uint4 q[2]; };

// Use the Tensor Data Mover when the toolchain exposes it (device pass only).
#if defined(__HIP_DEVICE_COMPILE__) && __has_builtin(__builtin_amdgcn_tensor_load_to_lds) && __has_builtin(__builtin_amdgcn_s_wait_tensorcnt)
#define ATTN_TDM 1
#else
#define ATTN_TDM 0
#endif

static __device__ __forceinline__ float bf2f(unsigned short h) {
  union { unsigned int u; float f; } c; c.u = ((unsigned int)h) << 16; return c.f;
}
static __device__ __forceinline__ unsigned short f2bf(float f) {
  union { float f; unsigned int u; } c; c.f = f;
  unsigned int u = c.u + 0x7FFFu + ((c.u >> 16) & 1u);   // round-to-nearest-even
  return (unsigned short)(u >> 16);
}
static __device__ __forceinline__ float sigf(float x) { return 1.0f / (1.0f + __expf(-x)); }

#if ATTN_TDM
// ---------------------------------------------------------------------------
// Issue a TDM 2D tile load: 64 contiguous bf16 along K x 128 rows (stride K)
// from global into LDS (rows packed contiguously, 64 elements each).
// D# layout per CDNA5 ISA.
// ---------------------------------------------------------------------------
static __device__ __forceinline__ void tdm_issue_b_tile(
    unsigned short* ldsDst, const unsigned short* gsrc, int K)
{
  const unsigned int ldsOff = (unsigned int)(unsigned long long)ldsDst; // LDS byte offset
  const unsigned long long ga = (unsigned long long)gsrc;               // global byte addr
  v4u_ g0;
  g0[0] = 1u;                                               // count=1, user mode
  g0[1] = ldsOff;                                           // lds_addr
  g0[2] = (unsigned int)ga;                                 // global_addr[31:0]
  g0[3] = (unsigned int)((ga >> 32) & 0x01FFFFFFu) | (2u << 30); // addr[56:32] | type=2
  const unsigned long long s0 = (unsigned long long)(unsigned int)K;   // dim0 stride (elems)
  v8i_ g1;
  g1[0] = (int)0x10000u;                                    // data_size=1 (2 bytes)
  g1[1] = (int)(64u << 16);                                 // tensor_dim0 = 64
  g1[2] = (int)(128u << 16);                                // tensor_dim1 = 128 (lo)
  g1[3] = (int)(64u << 16);                                 // tile_dim0 = 64
  g1[4] = (int)128u;                                        // tile_dim1 = 128, tile_dim2 = 0
  g1[5] = (int)(unsigned int)(s0 & 0xFFFFFFFFu);            // dim0_stride[31:0]
  g1[6] = (int)(unsigned int)((s0 >> 32) & 0xFFFFu);        // dim0_stride[47:32]
  g1[7] = 0;                                                // dim1_stride (unused, 2D)
  v4i_ z4 = {0, 0, 0, 0};
#if __clang_major__ >= 23
  v8i_ z8 = {0, 0, 0, 0, 0, 0, 0, 0};
  __builtin_amdgcn_tensor_load_to_lds(g0, g1, z4, z4, z8, 0);
#else
  __builtin_amdgcn_tensor_load_to_lds(g0, g1, z4, z4, 0);
#endif
}

// ---------------------------------------------------------------------------
// TDM-staged K-accumulation: wave 0 DMA-streams 128x64 bf16 weight tiles into
// double-buffered LDS (TENSORcnt-gated), all 8 waves consume via ds_load_b128;
// A fragments stream directly from global (overlapped with the tensor wait).
// All 8 B fragments of a chunk are loaded into distinct registers before the
// 8 WMMAs issue, so ds loads clause/pipeline instead of draining per-WMMA.
// K must be a multiple of 64.
// ---------------------------------------------------------------------------
static __device__ __forceinline__ void gemm_accum_tdm(
    v8f acc[4],
    const unsigned short* __restrict__ A, int lda,
    const unsigned short* __restrict__ Bt, int K,
    int rowA, int colBlock, int wave, int wn,
    int klo, int kb, int bn,
    unsigned short (*ldsB)[128 * 64])
{
  const unsigned short* aRow = A + (size_t)rowA * lda;
  const int nC = K >> 6;

  if (wave == 0) {
    tdm_issue_b_tile(&ldsB[0][0], Bt + (size_t)colBlock * K, K);
    if (nC > 1) tdm_issue_b_tile(&ldsB[1][0], Bt + (size_t)colBlock * K + 64, K);
  }

  for (int c = 0; c < nC; ++c) {
    const int k0 = c << 6;
    // A fragments for both 32-wide sub-chunks (global; overlaps tensor wait)
    FragAB a0, a1;
    a0.q[0] = *(const uint4*)(aRow + k0 + klo);
    a0.q[1] = *(const uint4*)(aRow + k0 + klo + 8);
    a1.q[0] = *(const uint4*)(aRow + k0 + 32 + klo);
    a1.q[1] = *(const uint4*)(aRow + k0 + 32 + klo + 8);

    if (wave == 0) {
      if (c + 1 < nC) __builtin_amdgcn_s_wait_tensorcnt(1);
      else            __builtin_amdgcn_s_wait_tensorcnt(0);
    }
    __syncthreads();

    const unsigned short* bufc = &ldsB[c & 1][0];
    // load all 8 B fragments into distinct registers (16 x ds_load_b128)
    FragAB b[8];
#pragma unroll
    for (int j = 0; j < 4; ++j) {
      const unsigned short* pb = bufc + (size_t)(wn * 64 + j * 16 + bn) * 64 + kb;
      b[2 * j].q[0]     = *(const uint4*)(pb);
      b[2 * j].q[1]     = *(const uint4*)(pb + 8);
      b[2 * j + 1].q[0] = *(const uint4*)(pb + 32);
      b[2 * j + 1].q[1] = *(const uint4*)(pb + 40);
    }
    // 8 back-to-back WMMAs on the chunk
#pragma unroll
    for (int j = 0; j < 4; ++j) {
      acc[j] = __builtin_amdgcn_wmma_f32_16x16x32_bf16(
          false, a0.v, false, b[2 * j].v, (short)0, acc[j], false, false);
      acc[j] = __builtin_amdgcn_wmma_f32_16x16x32_bf16(
          false, a1.v, false, b[2 * j + 1].v, (short)0, acc[j], false, false);
    }
    __syncthreads();   // all waves done reading buf[c&1] (ds results consumed by wmma)

    if (wave == 0 && c + 2 < nC)
      tdm_issue_b_tile(&ldsB[c & 1][0], Bt + (size_t)colBlock * K + (size_t)(c + 2) * 64, K);
  }
}
#endif // ATTN_TDM

// ---------------------------------------------------------------------------
// Fallback: software-pipelined K-accumulation with register ping-pong so the
// global loads for chunk k+32 / k+64 are in flight while WMMAs for k issue.
// ---------------------------------------------------------------------------
static __device__ __forceinline__ void gemm_accum_pipelined(
    v8f acc[4],
    const unsigned short* __restrict__ A, int lda,
    const unsigned short* __restrict__ Bt, int K,
    int rowA, int colBase, int klo, int kb, int bn)
{
  const unsigned short* aRow = A + (size_t)rowA * lda;
  const unsigned short* bCol[4];
#pragma unroll
  for (int j = 0; j < 4; ++j)
    bCol[j] = Bt + (size_t)(colBase + j * 16 + bn) * K + kb;

  FragAB a0, a1, b0[4], b1[4];
  a0.q[0] = *(const uint4*)(aRow + klo);
  a0.q[1] = *(const uint4*)(aRow + klo + 8);
#pragma unroll
  for (int j = 0; j < 4; ++j) {
    b0[j].q[0] = *(const uint4*)(bCol[j]);
    b0[j].q[1] = *(const uint4*)(bCol[j] + 8);
  }

  for (int k0 = 0; k0 < K; k0 += 64) {
    {
      const unsigned short* pa = aRow + k0 + 32 + klo;
      a1.q[0] = *(const uint4*)(pa);
      a1.q[1] = *(const uint4*)(pa + 8);
#pragma unroll
      for (int j = 0; j < 4; ++j) {
        const unsigned short* pb = bCol[j] + k0 + 32;
        b1[j].q[0] = *(const uint4*)(pb);
        b1[j].q[1] = *(const uint4*)(pb + 8);
      }
    }
#pragma unroll
    for (int j = 0; j < 4; ++j)
      acc[j] = __builtin_amdgcn_wmma_f32_16x16x32_bf16(
          false, a0.v, false, b0[j].v, (short)0, acc[j], false, false);

    if (k0 + 64 < K) {
      const unsigned short* pa = aRow + k0 + 64 + klo;
      a0.q[0] = *(const uint4*)(pa);
      a0.q[1] = *(const uint4*)(pa + 8);
#pragma unroll
      for (int j = 0; j < 4; ++j) {
        const unsigned short* pb = bCol[j] + k0 + 64;
        b0[j].q[0] = *(const uint4*)(pb);
        b0[j].q[1] = *(const uint4*)(pb + 8);
      }
    }
#pragma unroll
    for (int j = 0; j < 4; ++j)
      acc[j] = __builtin_amdgcn_wmma_f32_16x16x32_bf16(
          false, a1.v, false, b1[j].v, (short)0, acc[j], false, false);
  }
}

// ---------------------------------------------------------------------------
// WMMA GEMM: C[M,N] = act( A[M,K](bf16,lda) @ Bt[N,K]^T(bf16) [+ A2@B2t^T] + bias )
// block = 256 threads = 8 waves; each wave computes a 16x64 strip of C.
// Block tile: 64 (M) x 128 (N). M,N multiples of 64/128; K multiple of 64.
// MODE 0: f32 store. MODE 1: bf16 store of tanh(x+bias). MODE 2: f32 x+bias.
// MODE 3: f32 x+bias scattered to [B,T,1000] layout (row = t*128+b), col<1000.
// ---------------------------------------------------------------------------
template<int MODE, bool DUAL>
__global__ __launch_bounds__(256)
void gemm_bf16(const unsigned short* __restrict__ A, int lda,
               const unsigned short* __restrict__ Bt,
               const unsigned short* __restrict__ A2, int lda2,
               const unsigned short* __restrict__ B2t,
               const float* __restrict__ bias,
               float* __restrict__ Cf, unsigned short* __restrict__ Cbf,
               int K, int K2, int ldc)
{
  const int lane = threadIdx.x & 31;
  const int wave = threadIdx.x >> 5;
  const int wm = wave & 3, wn = wave >> 2;
  const int rowBase = blockIdx.y * 64 + wm * 16;
  const int colBase = blockIdx.x * 128 + wn * 64;
  const int ar  = lane & 15;          // A row within tile
  const int klo = (lane >> 4) * 16;   // A K-chunk select
  const int bn  = lane & 15;          // B col within tile
  const int kb  = (lane >> 4) * 16;   // B K-chunk select

  v8f acc[4];
#pragma unroll
  for (int j = 0; j < 4; ++j)
#pragma unroll
    for (int i = 0; i < 8; ++i) acc[j][i] = 0.0f;

#if ATTN_TDM
  __shared__ __align__(16) unsigned short ldsB[2][128 * 64];  // 2 x 16KB double buffer
  gemm_accum_tdm(acc, A, lda, Bt, K, rowBase + ar, blockIdx.x * 128,
                 wave, wn, klo, kb, bn, ldsB);
  if (DUAL)
    gemm_accum_tdm(acc, A2, lda2, B2t, K2, rowBase + ar, blockIdx.x * 128,
                   wave, wn, klo, kb, bn, ldsB);
#else
  gemm_accum_pipelined(acc, A, lda, Bt, K, rowBase + ar, colBase, klo, kb, bn);
  if (DUAL)
    gemm_accum_pipelined(acc, A2, lda2, B2t, K2, rowBase + ar, colBase, klo, kb, bn);
#endif

#pragma unroll
  for (int j = 0; j < 4; ++j) {
#pragma unroll
    for (int i = 0; i < 8; ++i) {
      const int row = rowBase + (lane >> 4) * 8 + i;
      const int col = colBase + j * 16 + bn;
      const float v = acc[j][i];
      if (MODE == 0) {
        Cf[(size_t)row * ldc + col] = v;
      } else if (MODE == 1) {
        Cbf[(size_t)row * ldc + col] = f2bf(tanhf(v + bias[col]));
      } else if (MODE == 2) {
        Cf[(size_t)row * ldc + col] = v + bias[col];
      } else {
        if (col < 1000) {
          const int b = row & 127, t = row >> 7;
          Cf[(size_t)b * 48 * 1000 + (size_t)t * 1000 + col] = v + bias[col];
        }
      }
    }
  }
}

// ---------------------------------------------------------------------------
// Fused attention: scores = H[b] . qp[b], softmax over S, ctx = alpha . H[b].
// One 256-thread workgroup per batch row; grid.y selects head descriptor.
// ---------------------------------------------------------------------------
struct HeadDesc {
  const unsigned short* H;   // [B,S,K] bf16
  const float* qp;           // projected query slice base, stride qpStride per b
  unsigned short* dst;       // bf16 context dest (already offset to slice)
  unsigned short* dst2;      // optional second dest (nullable)
  int S, K, qpStride, dstStride, dst2Stride;
};
struct HeadPack { HeadDesc h[6]; };

__global__ __launch_bounds__(256)
void attend_fused(HeadPack hp)
{
  __shared__ float sAl[128];
  const HeadDesc d = hp.h[blockIdx.y];
  const int b = blockIdx.x;
  const int lane = threadIdx.x & 31, wave = threadIdx.x >> 5;
  const unsigned short* Hb = d.H + (size_t)b * d.S * d.K;
  const float* qp = d.qp + (size_t)b * d.qpStride;

  // scores
  for (int s = wave; s < d.S; s += 8) {
    const unsigned short* hr = Hb + (size_t)s * d.K;
    float acc = 0.0f;
    for (int k = lane; k < d.K; k += 32) acc += bf2f(hr[k]) * qp[k];
    for (int off = 16; off; off >>= 1) acc += __shfl_xor(acc, off, 32);
    if (lane == 0) sAl[s] = acc;
  }
  __syncthreads();

  // softmax (wave 0)
  if (threadIdx.x < 32) {
    float m = -3.4e38f;
    for (int s = lane; s < d.S; s += 32) m = fmaxf(m, sAl[s]);
    for (int off = 16; off; off >>= 1) m = fmaxf(m, __shfl_xor(m, off, 32));
    float e[4]; int idx = 0; float tot = 0.0f;
    for (int s = lane; s < d.S; s += 32) { float ev = __expf(sAl[s] - m); e[idx++] = ev; tot += ev; }
    for (int off = 16; off; off >>= 1) tot += __shfl_xor(tot, off, 32);
    const float inv = 1.0f / tot;
    idx = 0;
    for (int s = lane; s < d.S; s += 32) sAl[s] = e[idx++] * inv;
  }
  __syncthreads();

  // context
  for (int k = threadIdx.x; k < d.K; k += 256) {
    float acc = 0.0f;
    for (int s = 0; s < d.S; ++s) acc += sAl[s] * bf2f(Hb[(size_t)s * d.K + k]);
    const unsigned short hv = f2bf(acc);
    d.dst[(size_t)b * d.dstStride + k] = hv;
    if (d.dst2) d.dst2[(size_t)b * d.dst2Stride + k] = hv;
  }
}

// ---------------------------------------------------------------------------
// Small helper kernels
// ---------------------------------------------------------------------------
__global__ void cvt_f32_bf16_k(const float* __restrict__ src, unsigned short* __restrict__ dst, int n) {
  int i = blockIdx.x * 256 + threadIdx.x;
  if (i < n) dst[i] = f2bf(src[i]);
}
__global__ void cvt_pad_rows_k(const float* __restrict__ src, unsigned short* __restrict__ dst,
                               int rows, int cols, int validRows) {
  int i = blockIdx.x * 256 + threadIdx.x;
  if (i < rows * cols) {
    int r = i / cols;
    dst[i] = (r < validRows) ? f2bf(src[i]) : (unsigned short)0;
  }
}
__global__ void embed_gather_k(const float* __restrict__ emb, const int* __restrict__ actions,
                               unsigned short* __restrict__ Xbf) {
  int i = blockIdx.x * 256 + threadIdx.x;            // [T,B,E] flat, E=256, B=128
  if (i >= 48 * 128 * 256) return;
  int e = i & 255, b = (i >> 8) & 127, t = i >> 15;
  int a = actions[b * 48 + t];
  Xbf[i] = f2bf(emb[(size_t)a * 256 + e]);
}
__global__ void bias_sum_k(const float* a, const float* b, float* o, int n) {
  int i = blockIdx.x * 256 + threadIdx.x;
  if (i < n) o[i] = a[i] + b[i];
}
__global__ void copy_xt_k(const unsigned short* __restrict__ Xbf, unsigned short* __restrict__ hk1, int t) {
  int i = blockIdx.x * 256 + threadIdx.x;            // 128*256
  if (i >= 128 * 256) return;
  int b = i >> 8, e = i & 255;
  hk1[(size_t)b * 1792 + 1536 + e] = Xbf[(size_t)t * 32768 + i];
}
// LSTM cell elementwise. Also stages x_{t+1} into hk1 for the next step so the
// per-step x-copy launch disappears from the sequential critical path.
__global__ void lstm_step_k(const float* __restrict__ gates, float* __restrict__ cst,
                            unsigned short* __restrict__ qbuf, unsigned short* __restrict__ hsbf,
                            const unsigned short* __restrict__ Xbf, unsigned short* __restrict__ hk1,
                            int t) {
  int i = blockIdx.x * 256 + threadIdx.x;            // 128*512
  if (i >= 128 * 512) return;
  int b = i >> 9, k = i & 511;
  const float* g = gates + (size_t)b * 2048;
  float ig = sigf(g[k]);
  float fg = sigf(g[512 + k]);
  float gg = tanhf(g[1024 + k]);
  float og = sigf(g[1536 + k]);
  float c = fg * cst[i] + ig * gg;
  cst[i] = c;
  float h = og * tanhf(c);
  unsigned short hb = f2bf(h);
  qbuf[(size_t)b * 1024 + k] = hb;                       // h slice of q=[h,z_c]
  hsbf[((size_t)t * 128 + b) * 512 + k] = hb;            // hs for output proj
  if (t + 1 < 48 && k < 256) {                           // stage x_{t+1}
    hk1[(size_t)b * 1792 + 1536 + k] = Xbf[(size_t)(t + 1) * 32768 + b * 256 + k];
  }
}

// ---------------------------------------------------------------------------
// Host: orchestrate the whole decoder on `stream`
// ---------------------------------------------------------------------------
extern "C" void kernel_launch(void* const* d_in, const int* in_sizes, int n_in,
                              void* d_out, int out_size, void* d_ws, size_t ws_size,
                              hipStream_t stream)
{
  (void)in_sizes; (void)n_in; (void)out_size; (void)ws_size;

  const float* ins      = (const float*)d_in[0];   // [128,64,512]
  const float* his      = (const float*)d_in[1];   // [128,128,512]
  const int*   actions  = (const int*)  d_in[2];   // [128,48]
  const float* cur      = (const float*)d_in[3];   // [128,96,128]
  const float* ini      = (const float*)d_in[4];   // [128,96,128]
  /* d_in[5] ins_valid: unused by reference */
  const float* emb      = (const float*)d_in[6];   // [1000,256]
  const float* W_c      = (const float*)d_in[7];   // [512,512]
  const float* W_p      = (const float*)d_in[8];   // [512,1024]
  const float* W_s_b_1  = (const float*)d_in[9];   // [128,1024]
  const float* W_s_b_2  = (const float*)d_in[10];
  const float* W_s_c_1  = (const float*)d_in[11];
  const float* W_s_c_2  = (const float*)d_in[12];
  const float* W_bd_w   = (const float*)d_in[13];  // [512,1792]
  const float* W_bd_b   = (const float*)d_in[14];  // [512]
  const float* W_ih     = (const float*)d_in[15];  // [2048,512]
  const float* W_hh     = (const float*)d_in[16];  // [2048,512]
  const float* b_ih     = (const float*)d_in[17];  // [2048]
  const float* b_hh     = (const float*)d_in[18];  // [2048]
  const float* h2a_w    = (const float*)d_in[19];  // [1000,512]
  const float* h2a_b    = (const float*)d_in[20];  // [1000]
  float* out            = (float*)d_out;           // [128,48,1000]

  char* ws = (char*)d_ws; size_t off = 0;
  auto alloc = [&](size_t bytes) -> void* {
    void* p = ws + off; off += (bytes + 255) & ~(size_t)255; return p;
  };
  unsigned short* insbf  = (unsigned short*)alloc((size_t)128 * 64 * 512 * 2);
  unsigned short* hisbf  = (unsigned short*)alloc((size_t)128 * 128 * 512 * 2);
  unsigned short* inibf  = (unsigned short*)alloc((size_t)128 * 96 * 128 * 2);
  unsigned short* curbf  = (unsigned short*)alloc((size_t)128 * 96 * 128 * 2);
  unsigned short* Xbf    = (unsigned short*)alloc((size_t)48 * 128 * 256 * 2);
  unsigned short* Wc_bf  = (unsigned short*)alloc((size_t)512 * 512 * 2);
  unsigned short* Wcat   = (unsigned short*)alloc((size_t)1024 * 1024 * 2);
  unsigned short* Wbd_bf = (unsigned short*)alloc((size_t)512 * 1792 * 2);
  unsigned short* Wih_bf = (unsigned short*)alloc((size_t)2048 * 512 * 2);
  unsigned short* Whh_bf = (unsigned short*)alloc((size_t)2048 * 512 * 2);
  unsigned short* Wout   = (unsigned short*)alloc((size_t)1024 * 512 * 2);
  float*          bsum   = (float*)alloc((size_t)2048 * 4);
  float*          qp_c   = (float*)alloc((size_t)128 * 512 * 4);
  float*          qp_all = (float*)alloc((size_t)128 * 1024 * 4);
  unsigned short* qbuf   = (unsigned short*)alloc((size_t)128 * 1024 * 2);  // q=[h|z_c] bf16
  unsigned short* hk1    = (unsigned short*)alloc((size_t)128 * 1792 * 2);
  unsigned short* hkbf   = (unsigned short*)alloc((size_t)128 * 512 * 2);
  float*          gates  = (float*)alloc((size_t)128 * 2048 * 4);
  float*          cst    = (float*)alloc((size_t)128 * 512 * 4);
  unsigned short* hsbf   = (unsigned short*)alloc((size_t)48 * 128 * 512 * 2);

  // h0 = 0 (bf16 zeros in qbuf), c0 = 0
  hipMemsetAsync(qbuf, 0, (size_t)128 * 1024 * 2, stream);
  hipMemsetAsync(cst, 0, (size_t)128 * 512 * 4, stream);

  auto CVT = [&](const float* s, unsigned short* d, int n) {
    cvt_f32_bf16_k<<<(n + 255) / 256, 256, 0, stream>>>(s, d, n);
  };
  CVT(ins, insbf, 128 * 64 * 512);
  CVT(his, hisbf, 128 * 128 * 512);
  CVT(ini, inibf, 128 * 96 * 128);
  CVT(cur, curbf, 128 * 96 * 128);
  CVT(W_c, Wc_bf, 512 * 512);
  CVT(W_bd_w, Wbd_bf, 512 * 1792);
  CVT(W_ih, Wih_bf, 2048 * 512);
  CVT(W_hh, Whh_bf, 2048 * 512);
  // concatenated query-projection weight [1024,1024] = [W_p;W_s_b_1;W_s_b_2;W_s_c_1;W_s_c_2]
  CVT(W_p,     Wcat,              512 * 1024);
  CVT(W_s_b_1, Wcat + 512 * 1024, 128 * 1024);
  CVT(W_s_b_2, Wcat + 640 * 1024, 128 * 1024);
  CVT(W_s_c_1, Wcat + 768 * 1024, 128 * 1024);
  CVT(W_s_c_2, Wcat + 896 * 1024, 128 * 1024);
  cvt_pad_rows_k<<<(1024 * 512 + 255) / 256, 256, 0, stream>>>(h2a_w, Wout, 1024, 512, 1000);
  bias_sum_k<<<(2048 + 255) / 256, 256, 0, stream>>>(b_ih, b_hh, bsum, 2048);
  embed_gather_k<<<(48 * 128 * 256 + 255) / 256, 256, 0, stream>>>(emb, actions, Xbf);
  // prefill x_0 slice of hk1
  copy_xt_k<<<128, 256, 0, stream>>>(Xbf, hk1, 0);

  for (int t = 0; t < 48; ++t) {
    // qp_c[b,k] = sum_q W_c[k,q] * h[b,q]   (A = h slice of qbuf, lda=1024)
    gemm_bf16<0, false><<<dim3(4, 2), 256, 0, stream>>>(
        qbuf, 1024, Wc_bf, nullptr, 0, nullptr, nullptr, qp_c, nullptr, 512, 0, 512);

    // z_c = attend(ins, qp_c) -> hk1[:,0:512] and qbuf[:,512:1024]
    HeadPack hp1{};
    hp1.h[0] = HeadDesc{ insbf, qp_c, hk1, qbuf + 512, 64, 512, 512, 1792, 1024 };
    attend_fused<<<dim3(128, 1), 256, 0, stream>>>(hp1);

    // qp_all = q @ Wcat^T : [128,1024]x[1024,1024]
    gemm_bf16<0, false><<<dim3(8, 2), 256, 0, stream>>>(
        qbuf, 1024, Wcat, nullptr, 0, nullptr, nullptr, qp_all, nullptr, 1024, 0, 1024);

    // z_p, z1a, z1b, zka, zkb
    HeadPack hp2{};
    hp2.h[0] = HeadDesc{ hisbf, qp_all,       hk1 + 512,  nullptr, 128, 512, 1024, 1792, 0 };
    hp2.h[1] = HeadDesc{ inibf, qp_all + 512, hk1 + 1024, nullptr,  96, 128, 1024, 1792, 0 };
    hp2.h[2] = HeadDesc{ inibf, qp_all + 640, hk1 + 1152, nullptr,  96, 128, 1024, 1792, 0 };
    hp2.h[3] = HeadDesc{ curbf, qp_all + 768, hk1 + 1280, nullptr,  96, 128, 1024, 1792, 0 };
    hp2.h[4] = HeadDesc{ curbf, qp_all + 896, hk1 + 1408, nullptr,  96, 128, 1024, 1792, 0 };
    attend_fused<<<dim3(128, 5), 256, 0, stream>>>(hp2);

    // hk = tanh(hk1 @ W_bd^T + b) -> bf16
    gemm_bf16<1, false><<<dim3(4, 2), 256, 0, stream>>>(
        hk1, 1792, Wbd_bf, nullptr, 0, nullptr, W_bd_b, nullptr, hkbf, 1792, 0, 512);

    // gates = hk @ W_ih^T + h @ W_hh^T + (b_ih+b_hh)
    gemm_bf16<2, true><<<dim3(16, 2), 256, 0, stream>>>(
        hkbf, 512, Wih_bf, qbuf, 1024, Whh_bf, bsum, gates, nullptr, 512, 512, 2048);

    // LSTM cell elementwise; writes h (bf16) into qbuf and hsbf, c (f32) into cst,
    // and stages x_{t+1} into hk1.
    lstm_step_k<<<256, 256, 0, stream>>>(gates, cst, qbuf, hsbf, Xbf, hk1, t);
  }

  // out[b,t,:] = hs @ h2a_w^T + h2a_b   (N padded 1000->1024, guard in store)
  gemm_bf16<3, false><<<dim3(8, 96), 256, 0, stream>>>(
      hsbf, 512, Wout, nullptr, 0, nullptr, h2a_b, out, nullptr, 512, 0, 1024);
}
